// SpatialGAT_8985071583729
// MI455X (gfx1250) — matrix operand order repaired and checked by reading, not compiled, not run on hardware
//
#include <hip/hip_runtime.h>
#include <hip/hip_bf16.h>
#include <cstdint>
#include <cstddef>

// Problem dims (fixed by reference)
#define NND   8192   // nodes
#define FIN   256    // input features
#define NH    4      // heads
#define HDIM  64     // head dim
#define FOUT  256    // NH*HDIM
#define SLOPE 0.2f

typedef __attribute__((ext_vector_type(16))) _Float16 v16h;
typedef __attribute__((ext_vector_type(8)))  _Float16 v8h;
typedef __attribute__((ext_vector_type(8)))  float    v8f;

static __device__ __forceinline__ v16h cat8(v8h lo, v8h hi) {
  return __builtin_shufflevector(lo, hi, 0,1,2,3,4,5,6,7,8,9,10,11,12,13,14,15);
}

// ---------------------------------------------------------------------------
// Kernel 0a: convert x (f32) -> xh (f16), row-major [NND, FIN]
// ---------------------------------------------------------------------------
__global__ __launch_bounds__(256) void k_cvt_x(const float* __restrict__ x,
                                               _Float16* __restrict__ xh) {
  int idx = (blockIdx.x * 256 + threadIdx.x) * 8;
  #pragma unroll
  for (int i = 0; i < 8; ++i) xh[idx + i] = (_Float16)x[idx + i];
}

// ---------------------------------------------------------------------------
// Kernel 0b: W [FIN, FOUT] f32 -> Wt [FOUT, FIN] f16 (transposed so a WMMA
// B-fragment column is 32 contiguous K values)
// ---------------------------------------------------------------------------
__global__ __launch_bounds__(256) void k_cvt_w(const float* __restrict__ W,
                                               _Float16* __restrict__ wt) {
  int c = blockIdx.x;   // output column 0..255
  int k = threadIdx.x;  // input feature 0..255
  wt[c * FIN + k] = (_Float16)W[k * FOUT + c];
}

// ---------------------------------------------------------------------------
// Kernel 1: h = x * W via V_WMMA_F32_16X16X32_F16.
// One wave computes one 16x16 tile of h; grid = (NND/16, FOUT/64), block=128.
// Output stored as f16 row-major [NND, FOUT].
// ---------------------------------------------------------------------------
__global__ __launch_bounds__(128) void k_feat_gemm(const _Float16* __restrict__ xh,
                                                   const _Float16* __restrict__ wt,
                                                   _Float16* __restrict__ h16) {
  const int lane = threadIdx.x & 31;
  const int wave = threadIdx.x >> 5;
  const int i0   = blockIdx.x * 16;
  const int n0   = (blockIdx.y * 4 + wave) * 16;
  const int hi16 = lane & 16;          // upper half-wave flag (0 or 16)
  const int m16  = lane & 15;
  const int row  = i0 + m16;           // A row for this lane
  const int col  = n0 + m16;           // B column for this lane

  v8f acc = {};
  #pragma unroll
  for (int kc = 0; kc < FIN; kc += 32) {
    // A fragment: 16-bit A 16x32 layout. e<8 -> K=kbase+e ; e>=8 -> K=kbase+8+e
    const int kbA = kc + (hi16 ? 8 : 0);
    v8h alo = *(const v8h*)(xh + (size_t)row * FIN + kbA);
    v8h ahi = *(const v8h*)(xh + (size_t)row * FIN + kbA + 16);
    v16h a  = cat8(alo, ahi);
    // B fragment: lane holds column `col`, same K order as A (e<8 -> kbase+e ; e>=8 -> kbase+8+e)
    const int kbB = kc + (hi16 ? 8 : 0);
    v8h blo = *(const v8h*)(wt + (size_t)col * FIN + kbB);
    v8h bhi = *(const v8h*)(wt + (size_t)col * FIN + kbB + 16);
    v16h b  = cat8(blo, bhi);
    acc = __builtin_amdgcn_wmma_f32_16x16x32_f16(false, a, false, b,
                                                 (short)0, acc, false, false);
  }
  // C/D layout: VGPR r -> row m = r + 8*(lane>=16), col = lane&15
  #pragma unroll
  for (int r = 0; r < 8; ++r) {
    int m = r + (hi16 >> 1);
    h16[(size_t)(i0 + m) * FOUT + col] = (_Float16)acc[r];
  }
}

// ---------------------------------------------------------------------------
// Kernel 2: attention logits src/dst [NND, NH] (f32) + transpose h into
// hT [FOUT, NND] f16 so aggregation B-fragments are contiguous.
// One wave per node, block = 128 (4 waves).
// ---------------------------------------------------------------------------
__global__ __launch_bounds__(128) void k_logits_tr(const _Float16* __restrict__ h16,
                                                   const float* __restrict__ a_src,
                                                   const float* __restrict__ a_dst,
                                                   float* __restrict__ src,
                                                   float* __restrict__ dst,
                                                   _Float16* __restrict__ hT) {
  const int lane = threadIdx.x & 31;
  const int wave = threadIdx.x >> 5;
  const int n    = blockIdx.x * 4 + wave;

  float v[8];
  #pragma unroll
  for (int k = 0; k < 8; ++k) {
    int c = lane + 32 * k;
    _Float16 hv = h16[(size_t)n * FOUT + c];
    v[k] = (float)hv;
    hT[(size_t)c * NND + n] = hv;
  }
  #pragma unroll
  for (int hd = 0; hd < NH; ++hd) {
    int c0 = lane + 64 * hd;       // columns of head hd covered by k=2hd
    int c1 = c0 + 32;              // and k=2hd+1
    float s = v[2 * hd] * a_src[c0] + v[2 * hd + 1] * a_src[c1];
    float d = v[2 * hd] * a_dst[c0] + v[2 * hd + 1] * a_dst[c1];
    #pragma unroll
    for (int o = 16; o >= 1; o >>= 1) {
      s += __shfl_xor(s, o);
      d += __shfl_xor(d, o);
    }
    if (lane == 0) {
      src[n * NH + hd] = s;
      dst[n * NH + hd] = d;
    }
  }
}

// ---------------------------------------------------------------------------
// Kernel 3: fused masked softmax + aggregation.
//   out[i, hd*64+d] = (1/Z_i) * sum_j adj(i,j)*exp(leaky(src_i+dst_j)) * h[j,hd,d]
// Logits are bounded (|e| ~ 6) so no max-shift is needed; Z accumulated
// alongside the WMMA P*V products.
// Block = 256 threads = 8 waves = 4 heads x 2 row-groups over a 32-row i-tile.
// Adjacency is compressed at staging time: each wave loads one 32-col row
// chunk and a single wave32 __ballot turns it into a 32-bit mask -> one LDS
// dword per row (16x less LDS traffic in the hot loop than staging raw ints).
// ---------------------------------------------------------------------------
__global__ __launch_bounds__(256) void k_attn_agg(const int* __restrict__ adj,
                                                  const float* __restrict__ srcg,
                                                  const float* __restrict__ dstg,
                                                  const _Float16* __restrict__ hT,
                                                  const float* __restrict__ bias,
                                                  float* __restrict__ out) {
  __shared__ unsigned int maskS[32];   // 1 bit per (row, j) of current chunk
  __shared__ float dstS[32 * NH];      // dst logits for current j-chunk

  const int t     = threadIdx.x;
  const int lane  = t & 31;
  const int wave  = t >> 5;
  const int hd    = wave & 3;          // head
  const int rg    = wave >> 2;         // row group (0/1)
  const int i0    = blockIdx.x * 32;
  const int i0r   = i0 + rg * 16;      // this wave's 16-row tile
  const int m16   = lane & 15;
  const int hi16  = lane & 16;
  const int kbase = hi16 ? 8 : 0;

  const float srcv = srcg[(size_t)(i0r + m16) * NH + hd];
  const _Float16* vbase = hT + (size_t)(hd * HDIM) * NND + (hi16 ? 8 : 0);

  v8f acc0 = {}, acc1 = {}, acc2 = {}, acc3 = {};
  float zacc = 0.f;

  for (int jc = 0; jc < NND; jc += 32) {
    // ---- stage adj tile as 32 bitmasks + dst chunk ----
    // Wave w, step q loads row (w + 8q), columns jc..jc+31 (one 128B line),
    // then a wave32 ballot compresses it to one dword.
    #pragma unroll
    for (int q = 0; q < 4; ++q) {
      int rr = wave + 8 * q;
      int av = adj[(size_t)(i0 + rr) * NND + jc + lane];
      unsigned long long bal = __ballot(av > 0);
      if (lane == 0) maskS[rr] = (unsigned int)bal;
    }
    if (t < 128) dstS[t] = dstg[(size_t)jc * NH + t];
    __syncthreads();

    // prefetch next adj chunk (one 128B cacheline per row) -> global_prefetch_b8
    if (jc + 32 < NND && wave == 0) {
      __builtin_prefetch(&adj[(size_t)(i0 + lane) * NND + jc + 32], 0, 3);
    }

    // ---- build P (A-matrix, 16x32 f16) per the ISA 16-bit A layout ----
    const unsigned int mrow = maskS[m16];   // conflict-free broadcast
    v16h a;
    #pragma unroll
    for (int e = 0; e < 16; ++e) {
      int joff = kbase + e + ((e >= 8) ? 8 : 0);   // K index within chunk
      float w = 0.f;
      if ((mrow >> joff) & 1u) {
        float lt = srcv + dstS[joff * NH + hd];
        lt = (lt > 0.f) ? lt : SLOPE * lt;
        w = __expf(lt);
      }
      zacc += w;
      a[e] = (_Float16)w;
    }

    // ---- 4 WMMA tiles covering the 64 output dims of this head ----
    const _Float16* vb = vbase + jc;
#define DO_TILE(ACC, TT)                                                       \
    {                                                                          \
      const _Float16* p = vb + (size_t)((TT) * 16 + m16) * NND;                \
      v16h b = cat8(*(const v8h*)p, *(const v8h*)(p + 16));                    \
      ACC = __builtin_amdgcn_wmma_f32_16x16x32_f16(false, a, false, b,         \
                                                   (short)0, ACC, false, false);\
    }
    DO_TILE(acc0, 0)
    DO_TILE(acc1, 1)
    DO_TILE(acc2, 2)
    DO_TILE(acc3, 3)
#undef DO_TILE

    __syncthreads();   // protect LDS before next chunk's staging
  }

  // ---- epilogue: combine half-wave Z partials, normalize, add bias ----
  // Row m's j-space is split between lanes m and m+16.
  float ztot = zacc + __shfl_xor(zacc, 16);   // lanes m and m+16 now hold Z[m]
  float zrow[8];
  #pragma unroll
  for (int r = 0; r < 8; ++r) {
    // C/D element r lives at row m = r + 8*(lane>=16); fetch Z[m] from lane m
    zrow[r] = __shfl(ztot, r + (hi16 >> 1));
  }
  #pragma unroll
  for (int r = 0; r < 8; ++r) {
    int m = r + (hi16 >> 1);
    float inv = 1.0f / (zrow[r] + 1e-30f);
    int colb = hd * HDIM + m16;
    size_t ob = (size_t)(i0r + m) * FOUT + colb;
    out[ob +  0] = acc0[r] * inv + bias[colb +  0];
    out[ob + 16] = acc1[r] * inv + bias[colb + 16];
    out[ob + 32] = acc2[r] * inv + bias[colb + 32];
    out[ob + 48] = acc3[r] * inv + bias[colb + 48];
  }
}

// ---------------------------------------------------------------------------
// Launch
// ---------------------------------------------------------------------------
extern "C" void kernel_launch(void* const* d_in, const int* in_sizes, int n_in,
                              void* d_out, int out_size, void* d_ws, size_t ws_size,
                              hipStream_t stream) {
  (void)in_sizes; (void)n_in; (void)out_size; (void)ws_size;

  const float* x     = (const float*)d_in[0];   // [8192, 256]
  const int*   adj   = (const int*)  d_in[1];   // [8192, 8192]
  const float* W     = (const float*)d_in[2];   // [256, 4, 64] == [256, 256]
  const float* a_src = (const float*)d_in[3];   // [4, 64]
  const float* a_dst = (const float*)d_in[4];   // [4, 64]
  const float* bias  = (const float*)d_in[5];   // [256]
  float*       out   = (float*)d_out;           // [8192, 256]

  // Workspace layout (all regions fully written before being read)
  char* ws = (char*)d_ws;
  size_t off = 0;
  _Float16* xh  = (_Float16*)(ws + off); off += (size_t)NND * FIN * 2;   // 4 MB
  _Float16* wt  = (_Float16*)(ws + off); off += (size_t)FOUT * FIN * 2;  // 128 KB
  _Float16* h16 = (_Float16*)(ws + off); off += (size_t)NND * FOUT * 2;  // 4 MB
  _Float16* hT  = (_Float16*)(ws + off); off += (size_t)FOUT * NND * 2;  // 4 MB
  float*    src = (float*)   (ws + off); off += (size_t)NND * NH * 4;    // 128 KB
  float*    dst = (float*)   (ws + off); off += (size_t)NND * NH * 4;    // 128 KB

  // 0) precision conversion + W transpose
  k_cvt_x<<<dim3((NND * FIN) / (256 * 8)), dim3(256), 0, stream>>>(x, xh);
  k_cvt_w<<<dim3(FOUT), dim3(FIN), 0, stream>>>(W, wt);

  // 1) h = x * W  (WMMA f16 -> f32)
  k_feat_gemm<<<dim3(NND / 16, FOUT / 64), dim3(128), 0, stream>>>(xh, wt, h16);

  // 2) src/dst logits + hT transpose
  k_logits_tr<<<dim3(NND / 4), dim3(128), 0, stream>>>(h16, a_src, a_dst,
                                                       src, dst, hT);

  // 3) fused masked-softmax attention + WMMA aggregation
  k_attn_agg<<<dim3(NND / 32), dim3(256), 0, stream>>>(adj, src, dst, hT,
                                                       bias, out);
}
